// GAT_14010183319955
// MI455X (gfx1250) — matrix operand order repaired and checked by reading, not compiled, not run on hardware
//
#include <hip/hip_runtime.h>
#include <hip/hip_bf16.h>
#include <stdint.h>

// ---------------- types / helpers ----------------
typedef __attribute__((ext_vector_type(8)))  __bf16 v8bf;
typedef __attribute__((ext_vector_type(16))) __bf16 v16bf;
typedef __attribute__((ext_vector_type(8)))  float  v8f;

#define LALPHA 0.2f
#define Nn     4096
#define NFEATc 512
#define NHIDc  64
#define NHEADSc 8
#define NSc    64
#define NEc    131072
#define NHHc   512
#define KINc   1152   // 2*(NHH+NS)

static __device__ __forceinline__ __bf16 f2bf(float f) {
  uint32_t u = __builtin_bit_cast(uint32_t, f);
  u += 0x7FFFu + ((u >> 16) & 1u);            // round-to-nearest-even
  uint16_t h = (uint16_t)(u >> 16);
  return __builtin_bit_cast(__bf16, h);
}
static __device__ __forceinline__ float bf2f(__bf16 b) {
  uint16_t h = __builtin_bit_cast(uint16_t, b);
  uint32_t u = ((uint32_t)h) << 16;
  return __builtin_bit_cast(float, u);
}
static __device__ __forceinline__ v8bf pack8(v8f c) {
  v8bf r;
  #pragma unroll
  for (int i = 0; i < 8; ++i) r[i] = f2bf(c[i]);
  return r;
}
// A fragment (16x32 bf16, row-major source): lane m=L&15, hi=L>>4
// elems 0..15 -> K = k0 + hi*16 + i
static __device__ __forceinline__ v16bf load_a_frag(const __bf16* row, int k0, int hi) {
  const int ka = hi * 16;
  v8bf lo = *(const v8bf*)(row + k0 + ka);
  v8bf hv = *(const v8bf*)(row + k0 + ka + 8);
  v16bf a;
  #pragma unroll
  for (int i = 0; i < 8; ++i) { a[i] = lo[i]; a[8 + i] = hv[i]; }
  return a;
}
static __device__ __forceinline__ v8f wmma_bf16(v16bf a, v16bf b, v8f c) {
  return __builtin_amdgcn_wmma_f32_16x16x32_bf16(false, a, false, b, (short)0, c,
                                                 false, false);
}

// ---------------- conversion kernels ----------------
__global__ void conv_f2bf_kernel(const float* __restrict__ in, __bf16* __restrict__ out, int n) {
  int t = blockIdx.x * blockDim.x + threadIdx.x;
  if (t < n) out[t] = f2bf(in[t]);
}
// W_heads [8][512][64] -> WconcatT [n=512][k=512], n = h*64+j
__global__ void conv_wheads_kernel(const float* __restrict__ Wh, __bf16* __restrict__ o) {
  int t = blockIdx.x * blockDim.x + threadIdx.x;
  if (t >= 512 * 512) return;
  int n = t >> 9, k = t & 511;
  int h = n >> 6, j = n & 63;
  o[t] = f2bf(Wh[(size_t)h * 512 * 64 + (size_t)k * 64 + j]);
}
// W_out [k=512][n=512] -> WoutT [n][k]
__global__ void conv_woutT_kernel(const float* __restrict__ W, __bf16* __restrict__ o) {
  int t = blockIdx.x * blockDim.x + threadIdx.x;
  if (t >= 512 * 512) return;
  int n = t >> 9, k = t & 511;
  o[t] = f2bf(W[(size_t)k * 512 + n]);
}
// s [4096][64] -> hfin[:,512:576] bf16
__global__ void sfill_kernel(const float* __restrict__ s, __bf16* __restrict__ hfin) {
  int t = blockIdx.x * blockDim.x + threadIdx.x;
  if (t >= Nn * NSc) return;
  int i = t / NSc, k = t - i * NSc;
  hfin[(size_t)i * 576 + 512 + k] = f2bf(s[t]);
}

// ------- plain GEMM: C^T = (A @ B)^T, A row-major, B given as B^T -------
// 2x1 m-blocking per wave: two A fragments share each B fragment.
__global__ __launch_bounds__(256) void gemm_bt_ct(
    const __bf16* __restrict__ A, int lda,
    const __bf16* __restrict__ BT, int ldb,
    __bf16* __restrict__ CT, int ldct,
    int Mp, int Nt, int K) {   // Mp = m-tile PAIRS (M/32), Nt = n-tiles
  int wave = threadIdx.x >> 5, lane = threadIdx.x & 31;
  int l15 = lane & 15, hi = lane >> 4;
  int tile = blockIdx.x * 8 + wave;
  if (tile >= Mp * Nt) return;                 // wave-uniform
  int mp = tile % Mp, nt = tile / Mp;
  int mBase = mp * 32;
  const __bf16* arow0 = A + (size_t)(mBase + l15) * lda;
  const __bf16* arow1 = arow0 + (size_t)16 * lda;
  const __bf16* brow  = BT + (size_t)(nt * 16 + l15) * ldb;
  v8f c0 = {}, c1 = {};
  for (int k0 = 0; k0 < K; k0 += 32) {
    v16bf bf  = *(const v16bf*)(brow + k0 + hi * 16);
    v16bf af0 = load_a_frag(arow0, k0, hi);
    v16bf af1 = load_a_frag(arow1, k0, hi);
    c0 = wmma_bf16(af0, bf, c0);
    c1 = wmma_bf16(af1, bf, c1);
  }
  // C fragment: lane col n=l15, rows m = r + hi*8  -> contiguous in C^T
  __bf16* cp = CT + (size_t)(nt * 16 + l15) * ldct + mBase + hi * 8;
  *(v8bf*)cp = pack8(c0);
  *(v8bf*)(cp + 16) = pack8(c1);
}

// ---------------- per-node src/dst attention scalars ----------------
__global__ void srcdst_kernel(const __bf16* __restrict__ whT,  // [nh*fDim][4096]
                              const float* __restrict__ a,     // [nh][2*fDim]
                              float* __restrict__ srcOut,      // [nh][4096]
                              float* __restrict__ dstT,        // [4096][dstride]
                              int dstride, int fDim, int nh) {
  int t = blockIdx.x * blockDim.x + threadIdx.x;
  if (t >= nh * Nn) return;
  int h = t >> 12, i = t & (Nn - 1);
  const float* ah = a + (size_t)h * 2 * fDim;
  float s = 0.f, d = 0.f;
  for (int j = 0; j < fDim; ++j) {
    float w = bf2f(whT[(size_t)(h * fDim + j) * Nn + i]);
    s += w * ah[j];
    d += w * ah[fDim + j];
  }
  srcOut[h * Nn + i] = s;
  dstT[(size_t)i * dstride + h] = d;
}

// ---------------- masked-softmax row stats (max, Z) ----------------
template <int NH>
__global__ __launch_bounds__(256) void rowstats_kernel(
    const int* __restrict__ adj, const float* __restrict__ src,
    const float* __restrict__ dstT, float* __restrict__ mOut,
    float* __restrict__ zOut, int dstride) {
  __shared__ float red[NH * 256];
  int i = blockIdx.x, tid = threadIdx.x;
  const int* arow = adj + (size_t)i * Nn;
  float sv[NH], mx[NH];
  #pragma unroll
  for (int h = 0; h < NH; ++h) { sv[h] = src[h * Nn + i]; mx[h] = -__builtin_inff(); }
  for (int j = tid; j < Nn; j += 256) {
    if (arow[j] > 0) {
      const float* dj = dstT + (size_t)j * dstride;
      #pragma unroll
      for (int h = 0; h < NH; ++h) {
        float e = sv[h] + dj[h];
        e = (e < 0.f) ? (LALPHA * e) : e;
        mx[h] = fmaxf(mx[h], e);
      }
    }
  }
  #pragma unroll
  for (int h = 0; h < NH; ++h) red[h * 256 + tid] = mx[h];
  __syncthreads();
  for (int s2 = 128; s2 > 0; s2 >>= 1) {
    if (tid < s2) {
      #pragma unroll
      for (int h = 0; h < NH; ++h)
        red[h * 256 + tid] = fmaxf(red[h * 256 + tid], red[h * 256 + tid + s2]);
    }
    __syncthreads();
  }
  float mh[NH];
  #pragma unroll
  for (int h = 0; h < NH; ++h) mh[h] = red[h * 256];
  __syncthreads();
  float zz[NH];
  #pragma unroll
  for (int h = 0; h < NH; ++h) zz[h] = 0.f;
  for (int j = tid; j < Nn; j += 256) {
    if (arow[j] > 0) {
      const float* dj = dstT + (size_t)j * dstride;
      #pragma unroll
      for (int h = 0; h < NH; ++h) {
        float e = sv[h] + dj[h];
        e = (e < 0.f) ? (LALPHA * e) : e;
        zz[h] += __expf(e - mh[h]);
      }
    }
  }
  #pragma unroll
  for (int h = 0; h < NH; ++h) red[h * 256 + tid] = zz[h];
  __syncthreads();
  for (int s2 = 128; s2 > 0; s2 >>= 1) {
    if (tid < s2) {
      #pragma unroll
      for (int h = 0; h < NH; ++h) red[h * 256 + tid] += red[h * 256 + tid + s2];
    }
    __syncthreads();
  }
  if (tid == 0) {
    #pragma unroll
    for (int h = 0; h < NH; ++h) {
      bool empty = !(mh[h] > -__builtin_inff());
      mOut[h * Nn + i] = empty ? 0.f : mh[h];
      zOut[h * Nn + i] = empty ? 0.f : red[h * 256];  // z==0 flags "uniform" row
    }
  }
}

// ---------------- fused attention GEMM: hp = softmax(mask(e)) @ Wh ----------------
static __device__ __forceinline__ float att_p(int av, float dstv, float sv, float mv,
                                              float invZ, bool uni) {
  if (uni) return 1.f / (float)Nn;             // all-masked row -> uniform softmax
  if (av <= 0) return 0.f;
  float e = sv + dstv;
  e = (e < 0.f) ? (LALPHA * e) : e;
  return __expf(e - mv) * invZ;
}

template <int NT, bool ELU, bool HEADSEL>
__global__ __launch_bounds__(256) void att_gemm_kernel(
    const int* __restrict__ adj,
    const float* __restrict__ src, const float* __restrict__ dstT, int dstride,
    const float* __restrict__ mrow, const float* __restrict__ zrow,
    const __bf16* __restrict__ BT,   // Wh^T [512][4096]
    __bf16* __restrict__ Cout, int ldc, int jobNBase) {
  int wave = threadIdx.x >> 5, lane = threadIdx.x & 31;
  int l15 = lane & 15, hi = lane >> 4;
  int job = blockIdx.y;
  int head = HEADSEL ? job : 0;
  int nBase = job * jobNBase;
  int mt = blockIdx.x * 8 + wave;
  int irow = mt * 16 + l15;
  float sv = src[head * Nn + irow];
  float mv = mrow[head * Nn + irow];
  float zv = zrow[head * Nn + irow];
  bool uni = !(zv > 0.f);
  float invZ = uni ? 0.f : (1.f / zv);
  const int* arow = adj + (size_t)irow * Nn;
  v8f c[NT];
  v8f zf = {};
  #pragma unroll
  for (int t = 0; t < NT; ++t) c[t] = zf;
  for (int k0 = 0; k0 < Nn; k0 += 32) {
    int ja = k0 + hi * 16, jb = k0 + hi * 16 + 8;
    v16bf af;
    #pragma unroll
    for (int t = 0; t < 8; ++t) {
      int j = ja + t;
      float d = dstT[(size_t)j * dstride + head];
      af[t] = f2bf(att_p(arow[j], d, sv, mv, invZ, uni));
    }
    #pragma unroll
    for (int t = 0; t < 8; ++t) {
      int j = jb + t;
      float d = dstT[(size_t)j * dstride + head];
      af[8 + t] = f2bf(att_p(arow[j], d, sv, mv, invZ, uni));
    }
    #pragma unroll
    for (int nt = 0; nt < NT; ++nt) {
      const __bf16* brow = BT + (size_t)(nBase + nt * 16 + l15) * Nn;
      v16bf bf = *(const v16bf*)(brow + k0 + hi * 16);
      c[nt] = wmma_bf16(af, bf, c[nt]);
    }
  }
  #pragma unroll
  for (int nt = 0; nt < NT; ++nt) {
    int col = nBase + nt * 16 + l15;
    #pragma unroll
    for (int r = 0; r < 8; ++r) {
      int row = mt * 16 + r + hi * 8;
      float v = c[nt][r];
      if (ELU) v = (v > 0.f) ? v : (__expf(v) - 1.f);
      Cout[(size_t)row * ldc + col] = f2bf(v);
    }
  }
}

// ---------------- fused edge MLP: y = relu(gather(hfin) @ W1^T + b1) . W2 ----------------
// k-outer loop: A fragment loaded from LDS once per k-tile, reused by 4 n-tiles.
__global__ __launch_bounds__(256) void edge_mlp_kernel(
    const int* __restrict__ ids,      // [E][2]
    const __bf16* __restrict__ hfin,  // [4096][576]
    const __bf16* __restrict__ w1,    // [512][1152] (natural [n][k] layout)
    const float* __restrict__ b1, const float* __restrict__ w2,
    float* __restrict__ part) {       // [E][8] per-wave partials
  __shared__ __align__(16) __bf16 sA[16 * KINc];
  int eBase = blockIdx.x * 16;
  // stage 16 gathered edge rows (src||dst) into LDS, 16B chunks
  for (int cidx = threadIdx.x; cidx < 16 * 2 * 72; cidx += 256) {
    int row = cidx / 144, rem = cidx - row * 144;
    int half = rem / 72, q = rem - half * 72;
    int node = ids[(eBase + row) * 2 + half];
    const uint4* sp = (const uint4*)(hfin + (size_t)node * 576) + q;
    uint4* dp = (uint4*)(sA + row * KINc + half * 576) + q;
    *dp = *sp;
  }
  __syncthreads();
  int wave = threadIdx.x >> 5, lane = threadIdx.x & 31;
  int l15 = lane & 15, hi = lane >> 4;
  const __bf16* arow = sA + l15 * KINc;
  // 8 waves x 4 n-tiles = 32 n-tiles = 512 columns
  const __bf16* brow[4];
  #pragma unroll
  for (int q = 0; q < 4; ++q)
    brow[q] = w1 + (size_t)((wave + q * 8) * 16 + l15) * KINc;
  v8f c[4];
  v8f zf = {};
  #pragma unroll
  for (int q = 0; q < 4; ++q) c[q] = zf;
  for (int k0 = 0; k0 < KINc; k0 += 32) {
    v16bf af = load_a_frag(arow, k0, hi);       // ds_load once per k-tile
    #pragma unroll
    for (int q = 0; q < 4; ++q) {
      v16bf bf = *(const v16bf*)(brow[q] + k0 + hi * 16);
      c[q] = wmma_bf16(af, bf, c[q]);
    }
  }
  float acc[8];
  #pragma unroll
  for (int r = 0; r < 8; ++r) acc[r] = 0.f;
  #pragma unroll
  for (int q = 0; q < 4; ++q) {
    int n = (wave + q * 8) * 16 + l15;
    float b1n = b1[n], w2n = w2[n];
    #pragma unroll
    for (int r = 0; r < 8; ++r) {
      float v = c[q][r] + b1n;
      v = (v > 0.f) ? v : 0.f;
      acc[r] += v * w2n;                        // fuse second GEMV
    }
  }
  #pragma unroll
  for (int r = 0; r < 8; ++r) {
    #pragma unroll
    for (int msk = 1; msk < 16; msk <<= 1)
      acc[r] += __shfl_xor(acc[r], msk, 16);    // reduce over 16 columns
  }
  if (l15 == 0) {
    #pragma unroll
    for (int r = 0; r < 8; ++r)
      part[(size_t)(eBase + hi * 8 + r) * 8 + wave] = acc[r];
  }
}

__global__ void edge_reduce_kernel(const float* __restrict__ part,
                                   const float* __restrict__ b2,
                                   float* __restrict__ out) {
  int e = blockIdx.x * blockDim.x + threadIdx.x;
  if (e >= NEc) return;
  float s = b2[0];
  #pragma unroll
  for (int w = 0; w < 8; ++w) s += part[(size_t)e * 8 + w];
  out[e] = s;
}

// ---------------- host launch ----------------
extern "C" void kernel_launch(void* const* d_in, const int* in_sizes, int n_in,
                              void* d_out, int out_size, void* d_ws, size_t ws_size,
                              hipStream_t stream) {
  (void)in_sizes; (void)n_in; (void)out_size; (void)ws_size;
  const float* s        = (const float*)d_in[0];
  const float* x        = (const float*)d_in[1];
  const int*   adj      = (const int*)d_in[2];
  const int*   ids      = (const int*)d_in[3];
  const float* W_heads  = (const float*)d_in[4];
  const float* a_heads  = (const float*)d_in[5];
  const float* W_out    = (const float*)d_in[6];
  const float* a_out    = (const float*)d_in[7];
  const float* W1       = (const float*)d_in[8];
  const float* b1       = (const float*)d_in[9];
  const float* W2       = (const float*)d_in[10];
  const float* b2       = (const float*)d_in[11];
  float* out = (float*)d_out;

  size_t off = 0;
  auto carve = [&](size_t bytes) -> void* {
    void* r = (char*)d_ws + off;
    off += (bytes + 255) & ~(size_t)255;
    return r;
  };
  __bf16* x_bf   = (__bf16*)carve((size_t)Nn * NFEATc * 2);
  __bf16* wconT  = (__bf16*)carve((size_t)512 * 512 * 2);
  __bf16* woutT  = (__bf16*)carve((size_t)512 * 512 * 2);
  __bf16* w1bf   = (__bf16*)carve((size_t)512 * KINc * 2);
  __bf16* wh_t   = (__bf16*)carve((size_t)512 * Nn * 2);
  __bf16* wh2_t  = (__bf16*)carve((size_t)512 * Nn * 2);
  __bf16* h_bf   = (__bf16*)carve((size_t)Nn * 512 * 2);
  __bf16* hfin   = (__bf16*)carve((size_t)Nn * 576 * 2);
  float* src_all = (float*)carve((size_t)8 * Nn * 4);
  float* dstT8   = (float*)carve((size_t)Nn * 8 * 4);
  float* m_all   = (float*)carve((size_t)8 * Nn * 4);
  float* z_all   = (float*)carve((size_t)8 * Nn * 4);
  float* src2    = (float*)carve((size_t)Nn * 4);
  float* dst2    = (float*)carve((size_t)Nn * 4);
  float* m2      = (float*)carve((size_t)Nn * 4);
  float* z2      = (float*)carve((size_t)Nn * 4);
  float* part    = (float*)carve((size_t)NEc * 8 * 4);

  // 0) precision conversions / layout transposes
  conv_f2bf_kernel<<<(Nn * NFEATc) / 256, 256, 0, stream>>>(x, x_bf, Nn * NFEATc);
  conv_wheads_kernel<<<(512 * 512) / 256, 256, 0, stream>>>(W_heads, wconT);
  conv_woutT_kernel<<<(512 * 512) / 256, 256, 0, stream>>>(W_out, woutT);
  conv_f2bf_kernel<<<(512 * KINc) / 256, 256, 0, stream>>>(W1, w1bf, 512 * KINc);
  sfill_kernel<<<(Nn * NSc) / 256, 256, 0, stream>>>(s, hfin);

  // 1) Wh (all 8 heads fused): wh_t = (x @ Wconcat)^T   (128 m-pairs x 32 n-tiles)
  gemm_bt_ct<<<(128 * 32) / 8, 256, 0, stream>>>(x_bf, NFEATc, wconT, 512,
                                                 wh_t, Nn, 128, 32, 512);
  // 2) per-head src/dst scalars; 3) masked-softmax row stats
  srcdst_kernel<<<(8 * Nn) / 256, 256, 0, stream>>>(wh_t, a_heads, src_all, dstT8,
                                                    8, NHIDc, 8);
  rowstats_kernel<8><<<Nn, 256, 0, stream>>>(adj, src_all, dstT8, m_all, z_all, 8);
  // 4) heads attention GEMM + ELU -> h [4096,512]
  att_gemm_kernel<4, true, true><<<dim3(32, 8), 256, 0, stream>>>(
      adj, src_all, dstT8, 8, m_all, z_all, wh_t, h_bf, 512, 64);
  // 5) Wh2 = (h @ W_out)^T
  gemm_bt_ct<<<(128 * 32) / 8, 256, 0, stream>>>(h_bf, 512, woutT, 512,
                                                 wh2_t, Nn, 128, 32, 512);
  // 6) out-layer attention scalars + stats
  srcdst_kernel<<<Nn / 256, 256, 0, stream>>>(wh2_t, a_out, src2, dst2, 1, NHHc, 1);
  rowstats_kernel<1><<<Nn, 256, 0, stream>>>(adj, src2, dst2, m2, z2, 1);
  // 7) out attention GEMM -> hfin[:, 0:512] (no ELU)
  att_gemm_kernel<8, false, false><<<dim3(32, 4), 256, 0, stream>>>(
      adj, src2, dst2, 1, m2, z2, wh2_t, hfin, 576, 128);
  // 8) fused edge MLP (gather + GEMM + bias + relu + W2 dot), deterministic partials
  edge_mlp_kernel<<<NEc / 16, 256, 0, stream>>>(ids, hfin, w1bf, b1, W2, part);
  edge_reduce_kernel<<<NEc / 256, 256, 0, stream>>>(part, b2, out);
}
